// SCARF_22084721836816
// MI455X (gfx1250) — matrix operand order repaired and checked by reading, not compiled, not run on hardware
//
#include <hip/hip_runtime.h>
#include <hip/hip_bf16.h>
#include <cstdint>
#include <cstddef>

// ---------------------------------------------------------------------------
// Problem constants (match reference)
// ---------------------------------------------------------------------------
#define BSZ   8192
#define MFEAT 512
#define EDIM  1024
#define CORRL 307
#define BNEPS 1e-5f

typedef __bf16 bf16;
typedef __attribute__((ext_vector_type(16))) __bf16 v16bf;
typedef __attribute__((ext_vector_type(8)))  float  v8f;
typedef __attribute__((ext_vector_type(4)))  int    v4i_;

// gfx1250 async global->LDS copy path (guarded: falls back to reg staging)
#if defined(__gfx1250__) && __has_builtin(__builtin_amdgcn_global_load_async_to_lds_b128)
#define HAVE_ASYNC_LDS 1
typedef __attribute__((address_space(1))) v4i_ g_v4i;   // global int4
typedef __attribute__((address_space(3))) v4i_ l_v4i;   // LDS int4
__device__ __forceinline__ void async_cp16(const bf16* g, bf16* l) {
    __builtin_amdgcn_global_load_async_to_lds_b128((g_v4i*)g, (l_v4i*)l, 0, 0);
}
__device__ __forceinline__ void async_wait0() {
#if __has_builtin(__builtin_amdgcn_s_wait_asynccnt)
    __builtin_amdgcn_s_wait_asynccnt(0);
#else
    asm volatile("s_wait_asynccnt 0x0" ::: "memory");
#endif
}
#else
#define HAVE_ASYNC_LDS 0
#endif

// ---------------------------------------------------------------------------
// Small helpers
// ---------------------------------------------------------------------------
__device__ __forceinline__ uint32_t hash32(uint32_t v) {
    v ^= v >> 16; v *= 0x7feb352du;
    v ^= v >> 15; v *= 0x846ca68bu;
    v ^= v >> 16;
    return v;
}

// f32 -> bf16 elementwise (for the clean input x)
__global__ __launch_bounds__(256)
void f32_to_bf16_kernel(const float* __restrict__ in, bf16* __restrict__ out, size_t n) {
    size_t i = (size_t)blockIdx.x * 256 + threadIdx.x;
    if (i < n) out[i] = (bf16)in[i];
}

// W[K][N] f32  ->  Wt[N][K] bf16 (transposed so GEMM B-tiles are row-contiguous)
__global__ __launch_bounds__(256)
void convert_wt_kernel(const float* __restrict__ W, bf16* __restrict__ Wt, int K, int N) {
    size_t i = (size_t)blockIdx.x * 256 + threadIdx.x;
    if (i >= (size_t)K * N) return;
    int k = (int)(i / N);
    int n = (int)(i % N);
    Wt[(size_t)n * K + k] = (bf16)W[i];
}

// ---------------------------------------------------------------------------
// Corruption: per row, replace exactly ~CORRL of MFEAT features with
// Uniform(low, high) samples. One wave32 per row, 16 features per lane.
// Exact-count mask via bisection on the 307th order statistic of per-element
// uniforms (wave reduction with __shfl_xor, wave32).
// ---------------------------------------------------------------------------
__global__ __launch_bounds__(256)
void corrupt_kernel(const float* __restrict__ x, const float* __restrict__ low,
                    const float* __restrict__ high, bf16* __restrict__ xc) {
    const int lane = threadIdx.x & 31;
    const int row  = blockIdx.x * 8 + (threadIdx.x >> 5);
    const int cbase = lane * 16;

    float r[16], u[16];
#pragma unroll
    for (int i = 0; i < 16; ++i) {
        uint32_t idx = (uint32_t)(row * MFEAT + cbase + i);
        r[i] = (float)(hash32(idx * 2u + 1u) >> 8) * (1.0f / 16777216.0f);
        u[i] = (float)(hash32(idx * 2u + 0x9E3779B9u) >> 8) * (1.0f / 16777216.0f);
    }

    float lo = 0.0f, hi = 1.0f;
    for (int it = 0; it < 24; ++it) {
        float mid = 0.5f * (lo + hi);
        int c = 0;
#pragma unroll
        for (int i = 0; i < 16; ++i) c += (r[i] <= mid) ? 1 : 0;
        for (int off = 16; off > 0; off >>= 1) c += __shfl_xor(c, off, 32);
        if (c >= CORRL) hi = mid; else lo = mid;
    }
    const float thr = hi;

#pragma unroll
    for (int i = 0; i < 16; ++i) {
        int col = cbase + i;
        float xr = low[col] + u[i] * (high[col] - low[col]);
        float v  = (r[i] <= thr) ? xr : x[(size_t)row * MFEAT + col];
        xc[(size_t)row * MFEAT + col] = (bf16)v;
    }
}

// ---------------------------------------------------------------------------
// GEMM: C[M,N] = A[M,K](bf16) * Wt[N,K]^T(bf16) + bias, f32 accumulate.
// Block tile 128x256, BK=32, 256 threads = 8 waves (2x4), wave tile 64x64
// (4x4 WMMA 16x16 tiles, 128 acc VGPRs). Double-buffered LDS (48 KB) so
// loads for tile k+1 overlap the WMMA of tile k; one barrier/k-step.
// Fragment lane layouts per CDNA5 ISA 7.12.2 (16-bit A 16x32, B 32x16).
// ---------------------------------------------------------------------------
template <typename OutT>
__global__ __launch_bounds__(256)
void gemm_bf16_wmma(const bf16* __restrict__ A, const bf16* __restrict__ Wt,
                    const float* __restrict__ bias, OutT* __restrict__ C,
                    int N, int K) {
    __shared__ __align__(16) bf16 sA[2][128][32];   // 16 KB
    __shared__ __align__(16) bf16 sB[2][256][32];   // 32 KB

    const int tid  = threadIdx.x;
    const int lane = tid & 31;
    const int wave = tid >> 5;
    const int wm   = wave >> 2;       // 0..1 : 64 rows each
    const int wn   = wave & 3;        // 0..3 : 64 cols each
    const int rowBase = blockIdx.y * 128;
    const int colBase = blockIdx.x * 256;

    // Loader mapping: A tile 128x32 -> thread t covers row t/2, half (t&1);
    //                 B tile 256x32 -> thread t covers full row t.
    const int aRow = tid >> 1;
    const int aK   = (tid & 1) * 16;
    const bf16* gA = A  + (size_t)(rowBase + aRow) * K + aK;
    const bf16* gB = Wt + (size_t)(colBase + tid) * K;

    v8f acc[4][4] = {};

#if HAVE_ASYNC_LDS
    auto stage = [&](int kt, int buf) {
        async_cp16(gA + kt,      &sA[buf][aRow][aK]);
        async_cp16(gA + kt + 8,  &sA[buf][aRow][aK + 8]);
        async_cp16(gB + kt,      &sB[buf][tid][0]);
        async_cp16(gB + kt + 8,  &sB[buf][tid][8]);
        async_cp16(gB + kt + 16, &sB[buf][tid][16]);
        async_cp16(gB + kt + 24, &sB[buf][tid][24]);
    };
    stage(0, 0);
    async_wait0();
    __syncthreads();
#else
    uint4 ra[2], rb[4];
    auto loadStage = [&](int kt) {
        ra[0] = *(const uint4*)(gA + kt);
        ra[1] = *(const uint4*)(gA + kt + 8);
        rb[0] = *(const uint4*)(gB + kt);
        rb[1] = *(const uint4*)(gB + kt + 8);
        rb[2] = *(const uint4*)(gB + kt + 16);
        rb[3] = *(const uint4*)(gB + kt + 24);
    };
    auto storeStage = [&](int buf) {
        *(uint4*)&sA[buf][aRow][aK]     = ra[0];
        *(uint4*)&sA[buf][aRow][aK + 8] = ra[1];
        *(uint4*)&sB[buf][tid][0]       = rb[0];
        *(uint4*)&sB[buf][tid][8]       = rb[1];
        *(uint4*)&sB[buf][tid][16]      = rb[2];
        *(uint4*)&sB[buf][tid][24]      = rb[3];
    };
    loadStage(0);
    storeStage(0);
    __syncthreads();
#endif

    int cur = 0;
    for (int kt = 0; kt < K; kt += 32) {
        const bool hasNext = (kt + 32) < K;
#if HAVE_ASYNC_LDS
        if (hasNext) stage(kt + 32, cur ^ 1);
#else
        if (hasNext) loadStage(kt + 32);
#endif
        if (kt + 64 < K) {
            __builtin_prefetch(gA + kt + 64, 0, 1);   // global_prefetch_b8
            __builtin_prefetch(gB + kt + 64, 0, 1);
        }

        union Frag { v16bf v; uint4 q[2]; };

        // B fragments: lane = N (mod 16); low lanes K=0..7,16..23, high K=8..15,24..31
        Frag bq[4];
        const int kbB = (lane >> 4) << 3;
#pragma unroll
        for (int nt = 0; nt < 4; ++nt) {
            const bf16* p = &sB[cur][wn * 64 + nt * 16 + (lane & 15)][kbB];
            bq[nt].q[0] = *(const uint4*)p;
            bq[nt].q[1] = *(const uint4*)(p + 16);
        }

        // A fragments: lane = M (mod 16); low lanes K=0..7,16..23, high K=8..15,24..31
        const int kbA = (lane >> 4) << 3;
#pragma unroll
        for (int mt = 0; mt < 4; ++mt) {
            Frag aq;
            const bf16* p = &sA[cur][wm * 64 + mt * 16 + (lane & 15)][kbA];
            aq.q[0] = *(const uint4*)p;
            aq.q[1] = *(const uint4*)(p + 16);
#pragma unroll
            for (int nt = 0; nt < 4; ++nt) {
                acc[mt][nt] = __builtin_amdgcn_wmma_f32_16x16x32_bf16(
                    false, aq.v, false, bq[nt].v, (short)0, acc[mt][nt],
                    false, false);
            }
        }

#if HAVE_ASYNC_LDS
        if (hasNext) async_wait0();          // buf[cur^1] LDS writes complete
#else
        if (hasNext) storeStage(cur ^ 1);    // safe: buf[cur^1] reads were >=1 barrier ago
#endif
        __syncthreads();                     // publish buf[cur^1]; guard buf[cur] rewrite
        cur ^= 1;
    }

    // Epilogue: C/D layout -> element (r, lane): M = r + 8*(lane>=16), N = lane%16
#pragma unroll
    for (int mt = 0; mt < 4; ++mt) {
#pragma unroll
        for (int nt = 0; nt < 4; ++nt) {
            const int col  = colBase + wn * 64 + nt * 16 + (lane & 15);
            const float bc = bias[col];
#pragma unroll
            for (int r = 0; r < 8; ++r) {
                const int row = rowBase + wm * 64 + mt * 16 + ((lane >> 4) << 3) + r;
                C[(size_t)row * N + col] = (OutT)(acc[mt][nt][r] + bc);
            }
        }
    }
}

// ---------------------------------------------------------------------------
// BatchNorm statistics: per-column mean and rsqrt(var+eps) over BSZ rows.
// 256 threads = 32 columns x 8 row-strips per block; grid = N/32.
// ---------------------------------------------------------------------------
__global__ __launch_bounds__(256)
void bn_stats_kernel(const float* __restrict__ Z, float* __restrict__ mu,
                     float* __restrict__ rinv, int rows, int N) {
    const int cl  = threadIdx.x & 31;
    const int col = blockIdx.x * 32 + cl;
    const int rt  = threadIdx.x >> 5;
    float s = 0.f, s2 = 0.f;
    for (int r = rt; r < rows; r += 8) {
        float v = Z[(size_t)r * N + col];
        s += v; s2 += v * v;
    }
    __shared__ float ss[8][32], sq[8][32];
    ss[rt][cl] = s;
    sq[rt][cl] = s2;
    __syncthreads();
    if (rt == 0) {
        float ts = 0.f, tq = 0.f;
#pragma unroll
        for (int i = 0; i < 8; ++i) { ts += ss[i][cl]; tq += sq[i][cl]; }
        float mean = ts / (float)rows;
        float var  = tq / (float)rows - mean * mean;
        mu[col]   = mean;
        rinv[col] = rsqrtf(var + BNEPS);
    }
}

// BN apply + ReLU + bf16 convert for next GEMM's A operand
__global__ __launch_bounds__(256)
void bn_apply_kernel(const float* __restrict__ Z, const float* __restrict__ mu,
                     const float* __restrict__ rinv, const float* __restrict__ g,
                     const float* __restrict__ be, bf16* __restrict__ out,
                     int N, size_t total) {
    size_t i = (size_t)blockIdx.x * 256 + threadIdx.x;
    if (i >= total) return;
    int col = (int)(i % (size_t)N);
    float y = (Z[i] - mu[col]) * rinv[col] * g[col] + be[col];
    out[i] = (bf16)fmaxf(y, 0.0f);
}

// ---------------------------------------------------------------------------
// Host orchestration
// ---------------------------------------------------------------------------
extern "C" void kernel_launch(void* const* d_in, const int* in_sizes, int n_in,
                              void* d_out, int out_size, void* d_ws, size_t ws_size,
                              hipStream_t stream) {
    (void)in_sizes; (void)n_in; (void)out_size; (void)ws_size;

    const float* x    = (const float*)d_in[0];
    const float* low  = (const float*)d_in[1];
    const float* high = (const float*)d_in[2];
    const float* Wsrc[6]  = { (const float*)d_in[3],  (const float*)d_in[5],
                              (const float*)d_in[7],  (const float*)d_in[9],
                              (const float*)d_in[17], (const float*)d_in[19] };
    const float* bias[6]  = { (const float*)d_in[4],  (const float*)d_in[6],
                              (const float*)d_in[8],  (const float*)d_in[10],
                              (const float*)d_in[18], (const float*)d_in[20] };
    const float* gArr[4]  = { (const float*)d_in[11], (const float*)d_in[13],
                              (const float*)d_in[15], (const float*)d_in[21] };
    const float* beArr[4] = { (const float*)d_in[12], (const float*)d_in[14],
                              (const float*)d_in[16], (const float*)d_in[22] };
    const int Ks[6] = { MFEAT, EDIM, EDIM, EDIM, EDIM, EDIM };

    // Workspace layout
    size_t off = 0;
    auto alloc = [&](size_t bytes) -> void* {
        void* p = (void*)((char*)d_ws + off);
        off += (bytes + 255) & ~(size_t)255;
        return p;
    };
    bf16* wt[6];
    for (int i = 0; i < 6; ++i) wt[i] = (bf16*)alloc((size_t)Ks[i] * EDIM * sizeof(bf16));
    bf16*  Abuf = (bf16*)alloc((size_t)BSZ * EDIM * sizeof(bf16));
    bf16*  Bbuf = (bf16*)alloc((size_t)BSZ * EDIM * sizeof(bf16));
    float* Z    = (float*)alloc((size_t)BSZ * EDIM * sizeof(float));
    float* mu   = (float*)alloc(EDIM * sizeof(float));
    float* rinv = (float*)alloc(EDIM * sizeof(float));

    // One-time per call: bf16 transposed weights
    for (int i = 0; i < 6; ++i) {
        int total = Ks[i] * EDIM;
        convert_wt_kernel<<<(total + 255) / 256, 256, 0, stream>>>(Wsrc[i], wt[i], Ks[i], EDIM);
    }

    const size_t actTotal = (size_t)BSZ * EDIM;
    auto run_tower = [&](float* outPtr) {
        bf16* cur = Abuf;
        bf16* nxt = Bbuf;
        int bnIdx = 0;
        for (int li = 0; li < 6; ++li) {
            const int Kcur = Ks[li];
            dim3 grid(EDIM / 256, BSZ / 128);
            if (li == 3) {
                // enc final linear: no BN/ReLU, feeds head -> write bf16 directly
                gemm_bf16_wmma<bf16><<<grid, 256, 0, stream>>>(cur, wt[li], bias[li],
                                                               nxt, EDIM, Kcur);
                bf16* t = cur; cur = nxt; nxt = t;
            } else if (li == 5) {
                // head final linear: f32 straight to output
                gemm_bf16_wmma<float><<<grid, 256, 0, stream>>>(cur, wt[li], bias[li],
                                                                outPtr, EDIM, Kcur);
            } else {
                gemm_bf16_wmma<float><<<grid, 256, 0, stream>>>(cur, wt[li], bias[li],
                                                                Z, EDIM, Kcur);
                bn_stats_kernel<<<EDIM / 32, 256, 0, stream>>>(Z, mu, rinv, BSZ, EDIM);
                bn_apply_kernel<<<(int)(actTotal / 256), 256, 0, stream>>>(
                    Z, mu, rinv, gArr[bnIdx], beArr[bnIdx], nxt, EDIM, actTotal);
                ++bnIdx;
                bf16* t = cur; cur = nxt; nxt = t;
            }
        }
    };

    float* out0 = (float*)d_out;
    float* out1 = out0 + (size_t)BSZ * EDIM;

    // Tower 1: clean x
    f32_to_bf16_kernel<<<(BSZ * MFEAT) / 256, 256, 0, stream>>>(x, Abuf, (size_t)BSZ * MFEAT);
    run_tower(out0);

    // Tower 2: corrupted x
    corrupt_kernel<<<BSZ / 8, 256, 0, stream>>>(x, low, high, Abuf);
    run_tower(out1);
}
